// MultiHeadAttention_dePosi_13142599926300
// MI455X (gfx1250) — matrix operand order repaired and checked by reading, not compiled, not run on hardware
//
#include <hip/hip_runtime.h>
#include <hip/hip_bf16.h>

typedef __attribute__((ext_vector_type(16))) __bf16 v16bf;
typedef __attribute__((ext_vector_type(8)))  float  v8f;

union Frag {
    v16bf          v;
    unsigned short us[16];
    unsigned int   u[8];
    uint4          q4[2];
};

__device__ __forceinline__ v8f zero8() {
    v8f v;
#pragma unroll
    for (int i = 0; i < 8; ++i) v[i] = 0.0f;
    return v;
}

__device__ __forceinline__ unsigned short f2bf(float f) {
    union { float f; unsigned u; } x;
    x.f = f;
    unsigned r = x.u + 0x7FFFu + ((x.u >> 16) & 1u);
    return (unsigned short)(r >> 16);
}

__device__ __forceinline__ unsigned pack2(float a, float b) {
    return (unsigned)f2bf(a) | ((unsigned)f2bf(b) << 16);
}

__device__ __forceinline__ v8f wmma_bf16(const Frag& a, const Frag& b, v8f c) {
    // D = A(16x32 bf16) * B(32x16 bf16) + C(16x16 f32)
    return __builtin_amdgcn_wmma_f32_16x16x32_bf16(
        false, a.v, false, b.v, (short)0, c, false, false);
}

__device__ __forceinline__ float posi_bias(float P0, float sp1, float sp2, float P3,
                                           int i, int j) {
    float d = fabsf((float)(i - j));
    float a = __expf(-sp1 * d) + __expf(-sp2 * d);
    if (i < j) a += P3;
    return P0 * a;
}

// ---------------------------------------------------------------------------
// GEMM: C[M=8192, N=512] = A[8192,512] * W[512(out),512(in)]^T + bias
// MODE 0: Q proj -> bf16 head-major [b*8+h][q][64]
// MODE 1: K proj -> bf16 head-major [b*8+h][k][64]
// MODE 2: V proj -> bf16 head-major TRANSPOSED [b*8+h][d][1024]
// MODE 3: fc: A is bf16, out f32 = C + bias + residual  (pre-LayerNorm)
// Block 128 threads = 4 waves; tile BM=64,BN=64,BK=32; wave does 32x32.
// ---------------------------------------------------------------------------
__device__ __forceinline__ void stage_f32_16(const float* __restrict__ src,
                                             unsigned short* __restrict__ dst) {
    const float4* f4 = (const float4*)src;
    unsigned pk[8];
#pragma unroll
    for (int c = 0; c < 4; ++c) {
        float4 f = f4[c];
        pk[2 * c]     = pack2(f.x, f.y);
        pk[2 * c + 1] = pack2(f.z, f.w);
    }
    ((uint4*)dst)[0] = make_uint4(pk[0], pk[1], pk[2], pk[3]);
    ((uint4*)dst)[1] = make_uint4(pk[4], pk[5], pk[6], pk[7]);
}

template <int MODE>
__global__ __launch_bounds__(128) void gemm_wmma(const void*  __restrict__ Ain,
                                                 const float* __restrict__ W,
                                                 const float* __restrict__ bias,
                                                 unsigned short* __restrict__ outb,
                                                 float* __restrict__ outf,
                                                 const float* __restrict__ resid) {
    __shared__ __align__(16) unsigned short As[64 * 40];
    __shared__ __align__(16) unsigned short Ws[64 * 40];
    const int tid = threadIdx.x, lane = tid & 31, w = tid >> 5;
    const int gm = blockIdx.x * 64, gn = blockIdx.y * 64;
    const int ln15 = lane & 15, hi8 = (lane & 16) >> 1, hi16 = lane & 16;
    const int srow = tid >> 1, skc = (tid & 1) * 16;

    v8f acc[2][2];
#pragma unroll
    for (int mt = 0; mt < 2; ++mt)
#pragma unroll
        for (int nt = 0; nt < 2; ++nt) acc[mt][nt] = zero8();

    for (int k0 = 0; k0 < 512; k0 += 32) {
        __syncthreads();
        if (MODE < 3) {
            const float* ap = (const float*)Ain + (size_t)(gm + srow) * 512 + k0 + skc;
            stage_f32_16(ap, &As[srow * 40 + skc]);
        } else {
            const unsigned short* ap =
                (const unsigned short*)Ain + (size_t)(gm + srow) * 512 + k0 + skc;
            ((uint4*)&As[srow * 40 + skc])[0] = ((const uint4*)ap)[0];
            ((uint4*)&As[srow * 40 + skc])[1] = ((const uint4*)ap)[1];
        }
        {
            const float* wp = W + (size_t)(gn + srow) * 512 + k0 + skc;
            stage_f32_16(wp, &Ws[srow * 40 + skc]);
        }
        __syncthreads();

        Frag af[2], bf2[2];
#pragma unroll
        for (int mt = 0; mt < 2; ++mt) {
            int row = (w & 1) * 32 + mt * 16 + ln15;
            af[mt].q4[0] = *(const uint4*)&As[row * 40 + hi8];       // K 0..7 (+8 hi)
            af[mt].q4[1] = *(const uint4*)&As[row * 40 + 16 + hi8];  // K 16..23 (+8 hi)
        }
#pragma unroll
        for (int nt = 0; nt < 2; ++nt) {
            int row = (w >> 1) * 32 + nt * 16 + ln15;
            bf2[nt].q4[0] = *(const uint4*)&Ws[row * 40 + hi8];      // K 0..7 (+8 hi)
            bf2[nt].q4[1] = *(const uint4*)&Ws[row * 40 + 16 + hi8];
        }
#pragma unroll
        for (int mt = 0; mt < 2; ++mt)
#pragma unroll
            for (int nt = 0; nt < 2; ++nt)
                acc[mt][nt] = wmma_bf16(af[mt], bf2[nt], acc[mt][nt]);
    }

#pragma unroll
    for (int mt = 0; mt < 2; ++mt)
#pragma unroll
        for (int nt = 0; nt < 2; ++nt) {
            int Mg = gm + (w & 1) * 32 + mt * 16 + hi8;
            int n  = gn + (w >> 1) * 32 + nt * 16 + ln15;
            float bn = bias[n];
#pragma unroll
            for (int r = 0; r < 8; ++r) {
                int m = Mg + r;
                float c = acc[mt][nt][r] + bn;
                if (MODE == 0 || MODE == 1) {
                    int bb = m >> 10, qq = m & 1023, hh = n >> 6, dd = n & 63;
                    outb[((size_t)(bb * 8 + hh) << 16) + ((size_t)qq << 6) + dd] = f2bf(c);
                } else if (MODE == 2) {
                    int bb = m >> 10, qq = m & 1023, hh = n >> 6, dd = n & 63;
                    outb[(((size_t)(bb * 8 + hh) * 64 + dd) << 10) + qq] = f2bf(c);
                } else {
                    size_t idx = (size_t)m * 512 + n;
                    outf[idx] = c + resid[idx];
                }
            }
        }
}

// ---------------------------------------------------------------------------
// Attention: per (b,h,16 q-rows per wave). S^T = K * Q^T so the softmaxed
// tile is directly the A-fragment of O = P * V (zero cross-lane movement).
// ---------------------------------------------------------------------------
__device__ __forceinline__ void score_step(const unsigned short* __restrict__ Kp,
                                           const Frag* qf, int j, int lane, int qi,
                                           float invT, float P0, float sp1, float sp2,
                                           float P3, float* sv) {
    const int ln15 = lane & 15, hi8 = (lane & 16) >> 1;
    v8f s1 = zero8(), s2 = zero8();
#pragma unroll
    for (int df = 0; df < 2; ++df) {
        Frag ka;
        {
            size_t base = (size_t)(j + ln15) * 64 + df * 32 + hi8;
            ka.q4[0] = *(const uint4*)&Kp[base];
            ka.q4[1] = *(const uint4*)&Kp[base + 16];
        }
        s1 = wmma_bf16(ka, qf[df], s1);
        {
            size_t base = (size_t)(j + 16 + ln15) * 64 + df * 32 + hi8;
            ka.q4[0] = *(const uint4*)&Kp[base];
            ka.q4[1] = *(const uint4*)&Kp[base + 16];
        }
        s2 = wmma_bf16(ka, qf[df], s2);
    }
#pragma unroll
    for (int r = 0; r < 8; ++r) {
        int j1 = j + hi8 + r;
        int j2 = j + 16 + hi8 + r;
        sv[r]     = s1[r] * invT + posi_bias(P0, sp1, sp2, P3, qi, j1);
        sv[8 + r] = s2[r] * invT + posi_bias(P0, sp1, sp2, P3, qi, j2);
    }
}

__global__ __launch_bounds__(128) void attn_wmma(
    const unsigned short* __restrict__ qh, const unsigned short* __restrict__ kh,
    const unsigned short* __restrict__ vt, const float* __restrict__ p0,
    const float* __restrict__ p1, const float* __restrict__ p2,
    const float* __restrict__ p3, unsigned short* __restrict__ o_bf,
    float* __restrict__ attn_out) {
    const int tid = threadIdx.x, lane = tid & 31, w = tid >> 5;
    const int bid = blockIdx.x, bh = bid >> 4, qblk = bid & 15;
    const int b = bh >> 3, h = bh & 7;
    const int ln15 = lane & 15, hi8 = (lane & 16) >> 1, hi16 = lane & 16;
    const size_t hoff = (size_t)(b * 8 + h) << 16;  // 1024*64 per head
    const unsigned short* Qp = qh + hoff;
    const unsigned short* Kp = kh + hoff;
    const unsigned short* Vp = vt + hoff;
    const int qbase = qblk * 64 + w * 16;
    const int qi = qbase + ln15;
    const float invT = 0.125f;
    const float P0 = p0[h], P3 = p3[h];
    const float sp1 = log1pf(__expf(p1[h]));
    const float sp2 = log1pf(__expf(p2[h]));

    Frag qf[2];  // Q^T B-fragments: lane=q, d-values hi8..+7 and 16+hi8..+7
#pragma unroll
    for (int df = 0; df < 2; ++df) {
        size_t base = (size_t)qi * 64 + df * 32 + hi8;
        qf[df].q4[0] = *(const uint4*)&Qp[base];
        qf[df].q4[1] = *(const uint4*)&Qp[base + 16];
    }

    v8f oacc[4];
#pragma unroll
    for (int t = 0; t < 4; ++t) oacc[t] = zero8();
    float rm = -1e30f, rs = 0.0f;

    // Pass 1: online softmax + O accumulation
    for (int j = 0; j < 1024; j += 32) {
        float sv[16];
        score_step(Kp, qf, j, lane, qi, invT, P0, sp1, sp2, P3, sv);
        float mloc = sv[0];
#pragma unroll
        for (int i = 1; i < 16; ++i) mloc = fmaxf(mloc, sv[i]);
        mloc = fmaxf(mloc, __shfl_xor(mloc, 16, 32));
        float rmn = fmaxf(rm, mloc);
        float sc = __expf(rm - rmn);
        float pv[16], psum = 0.0f;
#pragma unroll
        for (int i = 0; i < 16; ++i) {
            pv[i] = __expf(sv[i] - rmn);
            psum += pv[i];
        }
        rs = rs * sc + psum;
        rm = rmn;
        float sc8[8];
#pragma unroll
        for (int r = 0; r < 8; ++r) sc8[r] = __shfl(sc, hi8 + r, 32);
        Frag pf;  // P as A-fragment (M=q, K=32 keys), no shuffles needed
#pragma unroll
        for (int i = 0; i < 8; ++i)
            pf.u[i] = (unsigned)f2bf(pv[2 * i]) | ((unsigned)f2bf(pv[2 * i + 1]) << 16);
#pragma unroll
        for (int t = 0; t < 4; ++t) {
#pragma unroll
            for (int r = 0; r < 8; ++r) oacc[t][r] *= sc8[r];
            Frag vf;  // V B-fragment: lane=d, keys hi8..+7 and 16+hi8..+7 (V^T layout)
            int dcol = t * 16 + ln15;
            size_t vb = ((size_t)dcol << 10) + j + hi8;
            vf.q4[0] = *(const uint4*)&Vp[vb];
            vf.q4[1] = *(const uint4*)&Vp[vb + 16];
            oacc[t] = wmma_bf16(pf, vf, oacc[t]);
        }
    }

    float rst = rs + __shfl_xor(rs, 16, 32);
    float inv8[8];
#pragma unroll
    for (int r = 0; r < 8; ++r) inv8[r] = 1.0f / __shfl(rst, hi8 + r, 32);

#pragma unroll
    for (int t = 0; t < 4; ++t) {
        int dcol = t * 16 + ln15;
#pragma unroll
        for (int r = 0; r < 8; ++r) {
            int qq = qbase + hi8 + r;
            o_bf[(size_t)(b * 1024 + qq) * 512 + h * 64 + dcol] =
                f2bf(oacc[t][r] * inv8[r]);
        }
    }

    // Pass 2: recompute scores (L2-resident K), write normalized attn once
    float rinv = 1.0f / rst;
    size_t arow = ((size_t)(h * 8 + b) * 1024 + qi) * 1024;
    for (int j = 0; j < 1024; j += 32) {
        float sv[16];
        score_step(Kp, qf, j, lane, qi, invT, P0, sp1, sp2, P3, sv);
        float pw[16];
#pragma unroll
        for (int i = 0; i < 16; ++i) pw[i] = __expf(sv[i] - rm) * rinv;
        size_t o1 = arow + j + hi8;
        *(float4*)&attn_out[o1]     = make_float4(pw[0], pw[1], pw[2], pw[3]);
        *(float4*)&attn_out[o1 + 4] = make_float4(pw[4], pw[5], pw[6], pw[7]);
        size_t o2 = arow + j + 16 + hi8;
        *(float4*)&attn_out[o2]     = make_float4(pw[8], pw[9], pw[10], pw[11]);
        *(float4*)&attn_out[o2 + 4] = make_float4(pw[12], pw[13], pw[14], pw[15]);
    }
}

// ---------------------------------------------------------------------------
// In-place LayerNorm over rows of 512 (input already has bias+residual).
// ---------------------------------------------------------------------------
__global__ __launch_bounds__(256) void ln_kernel(float* __restrict__ io,
                                                 const float* __restrict__ g,
                                                 const float* __restrict__ be) {
    __shared__ float sb[8], sb2[8];
    const int t = threadIdx.x;
    float* p = io + (size_t)blockIdx.x * 512;
    float x0 = p[t], x1 = p[t + 256];
    float s = x0 + x1, ss = x0 * x0 + x1 * x1;
#pragma unroll
    for (int m = 16; m >= 1; m >>= 1) {
        s += __shfl_xor(s, m, 32);
        ss += __shfl_xor(ss, m, 32);
    }
    if ((t & 31) == 0) {
        sb[t >> 5] = s;
        sb2[t >> 5] = ss;
    }
    __syncthreads();
    float S = 0.0f, SS = 0.0f;
#pragma unroll
    for (int i = 0; i < 8; ++i) {
        S += sb[i];
        SS += sb2[i];
    }
    float mu = S * (1.0f / 512.0f);
    float var = SS * (1.0f / 512.0f) - mu * mu;
    float rstd = rsqrtf(var + 1e-5f);
    p[t]       = (x0 - mu) * rstd * g[t] + be[t];
    p[t + 256] = (x1 - mu) * rstd * g[t + 256] + be[t + 256];
}

extern "C" void kernel_launch(void* const* d_in, const int* in_sizes, int n_in,
                              void* d_out, int out_size, void* d_ws, size_t ws_size,
                              hipStream_t stream) {
    const float* q   = (const float*)d_in[0];
    const float* k   = (const float*)d_in[1];
    const float* v   = (const float*)d_in[2];
    const float* wqW = (const float*)d_in[3];
    const float* wqb = (const float*)d_in[4];
    const float* wkW = (const float*)d_in[5];
    const float* wkb = (const float*)d_in[6];
    const float* wvW = (const float*)d_in[7];
    const float* wvb = (const float*)d_in[8];
    const float* fcW = (const float*)d_in[9];
    const float* fcb = (const float*)d_in[10];
    const float* lng = (const float*)d_in[11];
    const float* lnb = (const float*)d_in[12];
    const float* p0  = (const float*)d_in[13];
    const float* p1  = (const float*)d_in[14];
    const float* p2  = (const float*)d_in[15];
    const float* p3  = (const float*)d_in[16];

    float* out0 = (float*)d_out;                              // [8,1024,512]
    float* attn = (float*)d_out + (size_t)8 * 1024 * 512;     // [64,1024,1024]

    char* ws = (char*)d_ws;
    unsigned short* qh  = (unsigned short*)(ws);               // 8 MB bf16
    unsigned short* khb = (unsigned short*)(ws + 8388608);     // 8 MB bf16
    unsigned short* vtb = (unsigned short*)(ws + 16777216);    // 8 MB bf16 (V^T)
    unsigned short* ob  = (unsigned short*)(ws + 25165824);    // 8 MB bf16

    dim3 g(128, 8), blk(128);
    gemm_wmma<0><<<g, blk, 0, stream>>>(q, wqW, wqb, qh, nullptr, nullptr);
    gemm_wmma<1><<<g, blk, 0, stream>>>(k, wkW, wkb, khb, nullptr, nullptr);
    gemm_wmma<2><<<g, blk, 0, stream>>>(v, wvW, wvb, vtb, nullptr, nullptr);
    attn_wmma<<<1024, 128, 0, stream>>>(qh, khb, vtb, p0, p1, p2, p3, ob, attn);
    gemm_wmma<3><<<g, blk, 0, stream>>>(ob, fcW, fcb, nullptr, out0, q);
    ln_kernel<<<8192, 256, 0, stream>>>(out0, lng, lnb);
}
